// CrossAttend_9474697855746
// MI455X (gfx1250) — matrix operand order repaired and checked by reading, not compiled, not run on hardware
//
#include <hip/hip_runtime.h>

// ---------------------------------------------------------------------------
// CrossAttend for MI455X (gfx1250, wave32, WMMA f32_16x16x32_f16)
// B=16, L=L2=1024, H=512
// v2: software-pipelined fragment loads (grouped loads before WMMA blocks)
// ---------------------------------------------------------------------------

#define HH   512
#define LQ   1024
#define NBAT 16
#define NEGV (-1000000000.0f)

typedef __attribute__((ext_vector_type(16))) _Float16 v16h;
typedef __attribute__((ext_vector_type(8)))  _Float16 v8h;
typedef __attribute__((ext_vector_type(8)))  float    v8f;

union HFrag { v16h v; v8h h[2]; };

// ---------------------------------------------------------------------------
// f32 -> f16 conversion (weights + activations)
// ---------------------------------------------------------------------------
__global__ void cvt_f32_f16(const float* __restrict__ in,
                            _Float16* __restrict__ out, int n) {
  int i = blockIdx.x * blockDim.x + threadIdx.x;
  int stride = gridDim.x * blockDim.x;
  for (; i < n; i += stride) out[i] = (_Float16)in[i];
}

// ---------------------------------------------------------------------------
// C[M,512] = A[M,512] (f16) @ W[512,512]^T (f16) + bias (f32), output f16.
// One wave computes a 16x64 tile. transposed=1 stores per-batch column-major
// ([B, H, L]) so attention's P*V B-operand is a contiguous load.
// ---------------------------------------------------------------------------
__global__ void gemm_proj(const _Float16* __restrict__ A,
                          const _Float16* __restrict__ W,
                          const float*    __restrict__ bias,
                          _Float16*       __restrict__ out,
                          int transposed) {
  const int lane  = threadIdx.x;
  const int half  = lane >> 4;      // which 16-lane half of the wave
  const int ln    = lane & 15;
  const int mbase = blockIdx.x * 16;
  const int nbase = blockIdx.y * 64;

  v8f z = {};
  v8f c0 = z, c1 = z, c2 = z, c3 = z;

  const _Float16* arow  = A + (size_t)(mbase + ln) * HH;
  const _Float16* wrow  = W + (size_t)(nbase + ln) * HH;

  // 16 k-steps in 4 groups of 4: issue all 24 loads of a group, then 16 WMMAs
#pragma unroll
  for (int g = 0; g < 4; ++g) {
    HFrag af[4];
    v16h  bf[4][4];
#pragma unroll
    for (int s = 0; s < 4; ++s) {
      const int h0 = (g * 4 + s) * 32;
      // A-frag (16x32, MxK): lane ln = row, K contiguous = 16*half + 0..15
      af[s].h[0] = *(const v8h*)(arow + h0 + 16 * half);
      af[s].h[1] = *(const v8h*)(arow + h0 + 16 * half + 8);
      // B-frags (32x16, KxN): lane = column n, K contiguous (32B)
      const _Float16* wb = wrow + h0 + 16 * half;
      bf[s][0] = *(const v16h*)(wb + (size_t)0  * HH);
      bf[s][1] = *(const v16h*)(wb + (size_t)16 * HH);
      bf[s][2] = *(const v16h*)(wb + (size_t)32 * HH);
      bf[s][3] = *(const v16h*)(wb + (size_t)48 * HH);
    }
#pragma unroll
    for (int s = 0; s < 4; ++s) {
      c0 = __builtin_amdgcn_wmma_f32_16x16x32_f16(false, af[s].v, false, bf[s][0], (short)0, c0, false, false);
      c1 = __builtin_amdgcn_wmma_f32_16x16x32_f16(false, af[s].v, false, bf[s][1], (short)0, c1, false, false);
      c2 = __builtin_amdgcn_wmma_f32_16x16x32_f16(false, af[s].v, false, bf[s][2], (short)0, c2, false, false);
      c3 = __builtin_amdgcn_wmma_f32_16x16x32_f16(false, af[s].v, false, bf[s][3], (short)0, c3, false, false);
    }
  }

  v8f acc[4] = {c0, c1, c2, c3};
#pragma unroll
  for (int t = 0; t < 4; ++t) {
    const int n = nbase + t * 16 + ln;
    const float bv = bias[n];
    if (!transposed) {
#pragma unroll
      for (int i = 0; i < 8; ++i) {
        const int m = mbase + i + 8 * half;        // C layout: row = vgpr + 8*half
        out[(size_t)m * HH + n] = (_Float16)(acc[t][i] + bv);
      }
    } else {
      v8h pk;
#pragma unroll
      for (int i = 0; i < 8; ++i) pk[i] = (_Float16)(acc[t][i] + bv);
      const int r  = mbase + 8 * half;             // 8 consecutive rows
      const int b  = r / LQ;
      const int rb = r % LQ;                       // 16-row tiles never cross batches
      *(v8h*)(out + (size_t)b * HH * LQ + (size_t)n * LQ + rb) = pk;
    }
  }
}

// ---------------------------------------------------------------------------
// Flash attention: one wave handles a 16-row query tile, streams keys in
// chunks of 32 with online softmax. Q:[B,LQ,H] f16, K:[B,LK,H] f16,
// VT:[B,H,LK] f16 (transposed), out:[B,LQ,H] f32.
// ---------------------------------------------------------------------------
__global__ void flash_attn(const _Float16* __restrict__ Q,
                           const _Float16* __restrict__ K,
                           const _Float16* __restrict__ VT,
                           float* __restrict__ out,
                           int LK, int maskdiag) {
  __shared__ _Float16 Qs[16 * HH];     // 16 KB: query tile
  __shared__ float    Sbuf[16 * 32];   // 2 KB: raw scores
  __shared__ _Float16 Pbuf[16 * 32];   // 1 KB: probabilities (f16)
  __shared__ float    mrow[16], lrow[16], arow[16];

  const int lane = threadIdx.x;
  const int half = lane >> 4;
  const int ln   = lane & 15;
  const int qb   = blockIdx.x * 16;    // query tile base (within batch)
  const int b    = blockIdx.y;
  const float scale = 0.044194173824159216f; // 1/sqrt(512)

  // Stage Q tile into LDS (row-major, matches global layout)
  {
    const v8h* qg = (const v8h*)(Q + ((size_t)b * LQ + qb) * HH);
    v8h* qs = (v8h*)Qs;
    for (int i = lane; i < 16 * HH / 8; i += 32) qs[i] = qg[i];
  }
  if (lane < 16) { mrow[lane] = -3.0e38f; lrow[lane] = 0.0f; }

  v8f z = {};
  v8f acc[32];                          // 16 x 512 f32 output accumulator
#pragma unroll
  for (int t = 0; t < 32; ++t) acc[t] = z;

  __syncthreads();

  const _Float16* kbase = K  + (size_t)b * LK * HH;
  const _Float16* vbase = VT + (size_t)b * HH * LK;

  for (int kb = 0; kb < LK; kb += 32) {
    // ---- S = (Q K^T) for 32 keys: two 16x16 tiles -------------------------
    // 16 k-steps in 8 groups of 2: group loads first, then 4 WMMAs
    v8f s0 = z, s1 = z;
    const _Float16* k0row = kbase + (size_t)(kb + ln)      * HH + 16 * half;
    const _Float16* k1row = kbase + (size_t)(kb + 16 + ln) * HH + 16 * half;
#pragma unroll
    for (int g = 0; g < 8; ++g) {
      HFrag af[2];
      v16h  b0[2], b1[2];
#pragma unroll
      for (int s = 0; s < 2; ++s) {
        const int h0 = (g * 2 + s) * 32;
        const _Float16* qrow = Qs + ln * HH + h0 + 16 * half;
        af[s].h[0] = *(const v8h*)(qrow);
        af[s].h[1] = *(const v8h*)(qrow + 8);
        b0[s] = *(const v16h*)(k0row + h0);
        b1[s] = *(const v16h*)(k1row + h0);
      }
#pragma unroll
      for (int s = 0; s < 2; ++s) {
        s0 = __builtin_amdgcn_wmma_f32_16x16x32_f16(false, af[s].v, false, b0[s], (short)0, s0, false, false);
        s1 = __builtin_amdgcn_wmma_f32_16x16x32_f16(false, af[s].v, false, b1[s], (short)0, s1, false, false);
      }
    }

    // ---- scale + diagonal mask, stage to LDS ------------------------------
#pragma unroll
    for (int i = 0; i < 8; ++i) {
      const int m = i + 8 * half;
      float x0 = s0[i] * scale;
      float x1 = s1[i] * scale;
      if (maskdiag) {
        if (qb + m == kb + ln)      x0 = NEGV;
        if (qb + m == kb + 16 + ln) x1 = NEGV;
      }
      Sbuf[m * 32 + ln]      = x0;
      Sbuf[m * 32 + 16 + ln] = x1;
    }
    __syncthreads();

    // ---- online softmax (one lane per query row) --------------------------
    if (lane < 16) {
      const float mp = mrow[lane];
      float mx = mp;
      for (int c = 0; c < 32; ++c) mx = fmaxf(mx, Sbuf[lane * 32 + c]);
      const float al = __expf(mp - mx);
      float sum = 0.0f;
      for (int c = 0; c < 32; ++c) {
        const float p = __expf(Sbuf[lane * 32 + c] - mx);
        sum += p;
        Pbuf[lane * 32 + c] = (_Float16)p;
      }
      mrow[lane] = mx;
      lrow[lane] = lrow[lane] * al + sum;
      arow[lane] = al;
    }
    __syncthreads();

    // ---- rescale accumulator, O += P * V ----------------------------------
    float av[8];
#pragma unroll
    for (int i = 0; i < 8; ++i) av[i] = arow[i + 8 * half];

    HFrag pf;                                        // P as A-operand (16x32)
    pf.h[0] = *(const v8h*)(Pbuf + ln * 32 + 16 * half);
    pf.h[1] = *(const v8h*)(Pbuf + ln * 32 + 16 * half + 8);

    // 32 h-tiles in 8 groups of 4: prefetch 4 V-fragments, then 4 WMMAs;
    // the alpha-rescale VALU work co-executes with the WMMA pipe.
#pragma unroll
    for (int tg = 0; tg < 8; ++tg) {
      v16h vf[4];
#pragma unroll
      for (int u = 0; u < 4; ++u) {
        const int t = tg * 4 + u;
        vf[u] = *(const v16h*)(vbase + (size_t)(t * 16 + ln) * LK + kb + 16 * half);
      }
#pragma unroll
      for (int u = 0; u < 4; ++u) {
        const int t = tg * 4 + u;
#pragma unroll
        for (int i = 0; i < 8; ++i) acc[t][i] *= av[i];
        acc[t] = __builtin_amdgcn_wmma_f32_16x16x32_f16(false, pf.v, false, vf[u], (short)0, acc[t], false, false);
      }
    }
    __syncthreads();   // protect Sbuf/Pbuf/arow before next chunk
  }

  // ---- normalize and store f32 output -------------------------------------
  float lv[8];
#pragma unroll
  for (int i = 0; i < 8; ++i) lv[i] = 1.0f / lrow[i + 8 * half];
  float* orow = out + ((size_t)b * LQ + qb) * HH;
#pragma unroll
  for (int t = 0; t < 32; ++t) {
#pragma unroll
    for (int i = 0; i < 8; ++i) {
      orow[(size_t)(i + 8 * half) * HH + t * 16 + ln] = acc[t][i] * lv[i];
    }
  }
}

// ---------------------------------------------------------------------------
// Host-side orchestration
// ---------------------------------------------------------------------------
extern "C" void kernel_launch(void* const* d_in, const int* in_sizes, int n_in,
                              void* d_out, int out_size, void* d_ws, size_t ws_size,
                              hipStream_t stream) {
  const float* q   = (const float*)d_in[0];
  const float* opp = (const float*)d_in[1];
  const float* Wq  = (const float*)d_in[2];
  const float* bq  = (const float*)d_in[3];
  const float* Wk  = (const float*)d_in[4];
  const float* bk  = (const float*)d_in[5];
  const float* Wv  = (const float*)d_in[6];
  const float* bv  = (const float*)d_in[7];
  float* out = (float*)d_out;

  const size_t NQ = (size_t)NBAT * LQ * HH;   // 8,388,608
  const size_t NW = (size_t)HH * HH;          // 262,144

  _Float16* w  = (_Float16*)d_ws;
  _Float16* qh = w;  w += NQ;   // q in f16
  _Float16* oh = w;  w += NQ;   // opp in f16
  _Float16* wq = w;  w += NW;
  _Float16* wk = w;  w += NW;
  _Float16* wv = w;  w += NW;
  _Float16* qp = w;  w += NQ;   // query projection  [B,L,H]
  _Float16* kk = w;  w += NQ;   // self keys         [B,L,H]
  _Float16* vt = w;  w += NQ;   // self values^T     [B,H,L]
  _Float16* ok = w;  w += NQ;   // opp keys          [B,L2,H]
  _Float16* ov = w;  w += NQ;   // opp values^T      [B,H,L2]

  cvt_f32_f16<<<2048, 256, 0, stream>>>(q,   qh, (int)NQ);
  cvt_f32_f16<<<2048, 256, 0, stream>>>(opp, oh, (int)NQ);
  cvt_f32_f16<<<256,  256, 0, stream>>>(Wq,  wq, (int)NW);
  cvt_f32_f16<<<256,  256, 0, stream>>>(Wk,  wk, (int)NW);
  cvt_f32_f16<<<256,  256, 0, stream>>>(Wv,  wv, (int)NW);

  dim3 gg(NBAT * LQ / 16, HH / 64);           // 1024 x 8 waves
  gemm_proj<<<gg, 32, 0, stream>>>(qh, wq, bq, qp, 0);   // qp = q Wq^T + bq
  gemm_proj<<<gg, 32, 0, stream>>>(qp, wk, bk, kk, 0);   // k  = qp Wk^T + bk
  gemm_proj<<<gg, 32, 0, stream>>>(qp, wv, bv, vt, 1);   // v^T
  gemm_proj<<<gg, 32, 0, stream>>>(oh, wk, bk, ok, 0);   // opp_k
  gemm_proj<<<gg, 32, 0, stream>>>(oh, wv, bv, ov, 1);   // opp_v^T

  dim3 ga(LQ / 16, NBAT);                     // 64 x 16 waves
  flash_attn<<<ga, 32, 0, stream>>>(qp, kk, vt, out,      LQ, 1);  // self (diag masked)
  flash_attn<<<ga, 32, 0, stream>>>(qp, ok, ov, out + NQ, LQ, 0);  // cross
}
